// CromLinearSOTA_63007170233050
// MI455X (gfx1250) — matrix operand order repaired and checked by reading, not compiled, not run on hardware
//
#include <hip/hip_runtime.h>
#include <hip/hip_bf16.h>

// Shapes from the reference:
//   x:        [4, 2048, 4096] fp32  -> M = 8192, K = 4096
//   codebook: [256, 8] fp16
//   indices:  [4096*4096/8] = 2,097,152 codes -> W [N=4096, K=4096] fp16 (row-major)
//   out = x_fp16 @ W.T  -> [8192, 4096] fp32
//
// Workspace layout (d_ws): [0,64MB) x_fp16 ; [64MB,96MB) W_fp16

typedef __attribute__((ext_vector_type(16))) _Float16 v16h;
typedef __attribute__((ext_vector_type(8)))  _Float16 v8h;
typedef __attribute__((ext_vector_type(8)))  float    v8f;
typedef __attribute__((ext_vector_type(4)))  float    v4f;
typedef __attribute__((ext_vector_type(4)))  int      v4i;

#define M_DIM 8192
#define N_DIM 4096
#define K_DIM 4096
#define N_CODES (N_DIM * K_DIM / 8)
#define BK 32
#define LDSS 40   // 32 halves + 8-half pad (80B rows) to spread LDS banks

// ---- CDNA5 async global->LDS copy (ASYNCcnt path), with portable fallback ----
#if __has_builtin(__builtin_amdgcn_global_load_async_to_lds_b128)
#define HAVE_ASYNC_LDS 1
typedef __attribute__((address_space(1))) v4i* as1_v4i_ptr;  // global int4*
typedef __attribute__((address_space(3))) v4i* as3_v4i_ptr;  // LDS int4*
static __device__ __forceinline__ void copy16_async(void* l, const void* g) {
  __builtin_amdgcn_global_load_async_to_lds_b128(
      (as1_v4i_ptr)(uintptr_t)g,
      (as3_v4i_ptr)(unsigned int)(uintptr_t)l,
      0, 0);
}
#if __has_builtin(__builtin_amdgcn_s_wait_asynccnt)
#define ASYNC_WAIT(n) __builtin_amdgcn_s_wait_asynccnt(n)
#else
#define ASYNC_WAIT(n) asm volatile("s_wait_asynccnt %0" ::"i"(n) : "memory")
#endif
#else
#define HAVE_ASYNC_LDS 0
static __device__ __forceinline__ void copy16_async(void* l, const void* g) {
  *(v8h*)l = *(const v8h*)g;   // global_load_b128 + ds_store_b128 fallback
}
#define ASYNC_WAIT(n)
#endif

// ---------------- fp32 -> fp16 conversion (x), 8 elems/thread ----------------
__global__ __launch_bounds__(256) void cvt_kernel(const float* __restrict__ x,
                                                  _Float16* __restrict__ xh) {
  size_t i = (size_t)blockIdx.x * blockDim.x + threadIdx.x;
  const v4f* p = (const v4f*)(x + i * 8);
  v4f a = p[0];
  v4f b = p[1];
  v8h o;
  o[0] = (_Float16)a[0]; o[1] = (_Float16)a[1];
  o[2] = (_Float16)a[2]; o[3] = (_Float16)a[3];
  o[4] = (_Float16)b[0]; o[5] = (_Float16)b[1];
  o[6] = (_Float16)b[2]; o[7] = (_Float16)b[3];
  *(v8h*)(xh + i * 8) = o;
}

// ---------------- codebook gather -> dense W, one 16B code/thread ----------------
__global__ __launch_bounds__(256) void dequant_kernel(const _Float16* __restrict__ cb,
                                                      const int* __restrict__ idx,
                                                      _Float16* __restrict__ W) {
  size_t i = (size_t)blockIdx.x * blockDim.x + threadIdx.x;
  int c = idx[i];
  v8h row = *(const v8h*)(cb + (size_t)c * 8);
  *(v8h*)(W + i * 8) = row;
}

// ---------------- WMMA GEMM: C[M,N] = A[M,K] * W[N,K]^T ----------------
// 256 threads = 8 wave32. Macro-tile 128(M) x 256(N), BK=32.
// Wave grid 2(M) x 4(N): each wave owns a 64x64 sub-tile = 4x4 WMMA frags
// -> 16 v_wmma per K-slice against 16 ds_load_b128 (1:1).
__global__ __launch_bounds__(256, 2) void gemm_kernel(const _Float16* __restrict__ A,
                                                      const _Float16* __restrict__ B,
                                                      float* __restrict__ C) {
  __shared__ __align__(16) _Float16 sA[2][128 * LDSS];  // 20 KB
  __shared__ __align__(16) _Float16 sB[2][256 * LDSS];  // 40 KB

  const int n0 = blockIdx.x * 256;
  const int m0 = blockIdx.y * 128;

  const int lane = threadIdx.x & 31;
  const int wv   = threadIdx.x >> 5;
  const int wm   = (wv & 1) * 64;   // wave M offset within tile
  const int wn   = (wv >> 1) * 64;  // wave N offset within tile
  const int lr   = lane & 15;
  const int hi   = lane >> 4;

  v8f acc[4][4] = {};

  // Stage one BK slice: A 128x32 halves (512 x 16B), B 256x32 halves (1024 x 16B)
  // = 6 async b128 copies per thread.
  auto stage = [&](int buf, int kk) {
    const int t = (int)threadIdx.x;
#pragma unroll
    for (int u = 0; u < 2; ++u) {
      const int c = t + u * 256;
      const int row = c >> 2, kc = (c & 3) * 8;
      copy16_async(&sA[buf][row * LDSS + kc],
                   A + (size_t)(m0 + row) * K_DIM + kk + kc);
    }
#pragma unroll
    for (int u = 0; u < 4; ++u) {
      const int c = t + u * 256;
      const int row = c >> 2, kc = (c & 3) * 8;
      copy16_async(&sB[buf][row * LDSS + kc],
                   B + (size_t)(n0 + row) * K_DIM + kk + kc);
    }
  };

  stage(0, 0);

  for (int kk = 0; kk < K_DIM; kk += BK) {
    const int buf = (kk >> 5) & 1;
    __syncthreads();                     // WAR: buffer buf^1 no longer being read
    if (kk + BK < K_DIM) {
      stage(buf ^ 1, kk + BK);
      ASYNC_WAIT(6);                     // in-order => previous group (buf) landed
    } else {
      ASYNC_WAIT(0);
    }
    __syncthreads();                     // all threads' group for buf landed

    // A fragments (16x32, k order this code uses with B: lanes 0-15 K0-15; lanes 16-31 K16-31, contiguous)
    v16h af[4];
#pragma unroll
    for (int mf = 0; mf < 4; ++mf) {
      const _Float16* p = &sA[buf][(wm + mf * 16 + lr) * LDSS + hi * 16];
      v8h lo  = *(const v8h*)p;
      v8h hi8 = *(const v8h*)(p + 8);
      af[mf] = __builtin_shufflevector(lo, hi8, 0, 1, 2, 3, 4, 5, 6, 7,
                                                8, 9, 10, 11, 12, 13, 14, 15);
    }
    // B fragments (32x16: lane n=lr, lanes 0-15 K0-15, lanes 16-31 K16-31, contiguous)
    v16h bf[4];
#pragma unroll
    for (int nf = 0; nf < 4; ++nf) {
      const _Float16* p = &sB[buf][(wn + nf * 16 + lr) * LDSS + hi * 16];
      v8h lo  = *(const v8h*)p;
      v8h hi8 = *(const v8h*)(p + 8);
      bf[nf] = __builtin_shufflevector(lo, hi8, 0, 1, 2, 3, 4, 5, 6, 7,
                                                8, 9, 10, 11, 12, 13, 14, 15);
    }

#pragma unroll
    for (int mf = 0; mf < 4; ++mf)
#pragma unroll
      for (int nf = 0; nf < 4; ++nf)
        acc[mf][nf] = __builtin_amdgcn_wmma_f32_16x16x32_f16(
            false, af[mf], false, bf[nf], (short)0, acc[mf][nf], false, false);
  }

  // Epilogue (32-bit C/D 16x16: VGPR r -> lanes 0-15 M=r, lanes 16-31 M=r+8; N=lane%16)
#pragma unroll
  for (int mf = 0; mf < 4; ++mf) {
#pragma unroll
    for (int nf = 0; nf < 4; ++nf) {
      const size_t col     = (size_t)(n0 + wn + nf * 16 + lr);
      const size_t rowbase = (size_t)(m0 + wm + mf * 16 + hi * 8);
#pragma unroll
      for (int r = 0; r < 8; ++r)
        C[(rowbase + r) * N_DIM + col] = acc[mf][nf][r];
    }
  }
}

extern "C" void kernel_launch(void* const* d_in, const int* in_sizes, int n_in,
                              void* d_out, int out_size, void* d_ws, size_t ws_size,
                              hipStream_t stream) {
  const float*    x   = (const float*)d_in[0];
  const _Float16* cb  = (const _Float16*)d_in[1];
  const int*      idx = (const int*)d_in[2];
  float*          out = (float*)d_out;

  _Float16* xh = (_Float16*)d_ws;                                      // 64 MB
  _Float16* W  = (_Float16*)((char*)d_ws + (size_t)M_DIM * K_DIM * 2); // +32 MB

  cvt_kernel<<<(M_DIM * K_DIM / 8) / 256, 256, 0, stream>>>(x, xh);
  dequant_kernel<<<N_CODES / 256, 256, 0, stream>>>(cb, idx, W);
  gemm_kernel<<<dim3(N_DIM / 256, M_DIM / 128), 256, 0, stream>>>(xh, W, out);
}
